// DoubleFusionV3_5626407158448
// MI455X (gfx1250) — matrix operand order repaired and checked by reading, not compiled, not run on hardware
//
#include <hip/hip_runtime.h>
#include <hip/hip_bf16.h>
#include <math.h>

// ---------------------------------------------------------------- types
typedef __attribute__((ext_vector_type(16))) __bf16        v16bf;
typedef __attribute__((ext_vector_type(8)))  float         v8f;
typedef __attribute__((ext_vector_type(8)))  unsigned int  v8u;
typedef __attribute__((ext_vector_type(4)))  int           v4i;

// CDNA5 async global->LDS staging (guarded: falls back to register bounce)
#if defined(__gfx1250__) && __has_builtin(__builtin_amdgcn_global_load_async_to_lds_b128)
#define GFX1250_ASYNC_LDS 1
#else
#define GFX1250_ASYNC_LDS 0
#endif

template <int N>
__device__ __forceinline__ void wait_asynccnt() {
#if __has_builtin(__builtin_amdgcn_s_wait_asynccnt)
  __builtin_amdgcn_s_wait_asynccnt((short)N);
#else
  if (N == 0) asm volatile("s_wait_asynccnt 0x0" ::: "memory");
  else        asm volatile("s_wait_asynccnt 0x1" ::: "memory");
#endif
}

// ---------------------------------------------------------------- dims
#define LQ   2048
#define BBS  2
#define EMB  1024
#define NH   32
#define HD   32
#define RS   (BBS*EMB)     // row stride of (L,B,E) tensors in elements
#define MROW (LQ*BBS)      // 4096 flattened rows

// ---------------------------------------------------------------- helpers
__device__ __forceinline__ unsigned short f2bf(float f) {
  unsigned u = __float_as_uint(f);
  u += 0x7FFFu + ((u >> 16) & 1u);           // round-to-nearest-even
  return (unsigned short)(u >> 16);
}
__device__ __forceinline__ v16bf mkv16(uint4 lo, uint4 hi) {
  v8u t;
  t[0]=lo.x; t[1]=lo.y; t[2]=lo.z; t[3]=lo.w;
  t[4]=hi.x; t[5]=hi.y; t[6]=hi.z; t[7]=hi.w;
  return __builtin_bit_cast(v16bf, t);
}
__device__ __forceinline__ v16bf mkv16u(v8u t) { return __builtin_bit_cast(v16bf, t); }
__device__ __forceinline__ float gelu_exact(float x) {
  return 0.5f * x * (1.0f + erff(x * 0.7071067811865476f));
}
__device__ __forceinline__ float silu(float x) {
  return x / (1.0f + __expf(-x));
}

// ---------------------------------------------------------------- f32 -> bf16
__global__ void cvt_f32_bf16(const float* __restrict__ in,
                             unsigned short* __restrict__ out, int n) {
  int i = blockIdx.x * blockDim.x + threadIdx.x;
  if (i < n) out[i] = f2bf(in[i]);
}

// ---------------------------------------------------------------- WMMA GEMM
// C[m,n] = act( sum_k X[m,k]*W[n,k] + bias[n] ) * (gate? silu(gate[m,n]) : 1)
// X: MxK bf16 row-major, W: NxK bf16 row-major (nn.Linear weight -> X @ W^T)
// 8 waves/block; wave owns 16 rows x 64 cols. The 64x32 W tile shared by all
// 8 waves is staged into double-buffered LDS with async global->LDS DMA
// (ASYNCcnt pipelined one k-step ahead), then consumed via ds_load_b128.

#if GFX1250_ASYNC_LDS
#define STAGE_W(buf, kk)                                                       \
  do {                                                                         \
    const unsigned short* _src = wsrc0 + (kk);                                 \
    unsigned short* _dst = &sW[(buf) * 2048 + srow * 32 + sseg];               \
    __builtin_amdgcn_global_load_async_to_lds_b128(                            \
        (__attribute__((address_space(1))) v4i*)_src,                          \
        (__attribute__((address_space(3))) v4i*)_dst, 0, 0);                   \
  } while (0)
#else
#define STAGE_W(buf, kk)                                                       \
  do {                                                                         \
    const unsigned short* _src = wsrc0 + (kk);                                 \
    unsigned short* _dst = &sW[(buf) * 2048 + srow * 32 + sseg];               \
    *(uint4*)_dst = *(const uint4*)_src;                                       \
  } while (0)
#endif

__global__ void gemm_wmma_bias_act(const unsigned short* __restrict__ X,
                                   const unsigned short* __restrict__ W,
                                   const float* __restrict__ bias,
                                   const float* __restrict__ gate,   // nullable MxN f32
                                   float* __restrict__ outF,         // nullable
                                   unsigned short* __restrict__ outB,// nullable
                                   int M, int N, int K, int act) {
  __shared__ __align__(32) unsigned short sW[2 * 64 * 32];  // 2 x 4KB W tiles

  const int lane = threadIdx.x & 31;
  const int wave = threadIdx.x >> 5;
  const int hi   = lane >> 4;     // 0/1 lane-half
  const int ln16 = lane & 15;
  const int m0 = blockIdx.x * 128 + wave * 16;
  const int n0 = blockIdx.y * 64;

  // staging role: 256 threads x 16B = one 64(row) x 32(k) bf16 tile
  const int srow = threadIdx.x >> 2;        // 0..63 (row of W tile)
  const int sseg = (threadIdx.x & 3) * 8;   // 8-elem (16B) segment in row
  const unsigned short* wsrc0 = W + (size_t)(n0 + srow) * K + sseg;

  v8f c[4] = {};
  const unsigned short* xrow = X + (size_t)(m0 + ln16) * K;

  STAGE_W(0, 0);

  for (int kk = 0; kk < K; kk += 32) {
    const int  cur  = (kk >> 5) & 1;
    const bool more = (kk + 32 < K);
    if (more) STAGE_W(cur ^ 1, kk + 32);     // DMA next tile while computing
#if GFX1250_ASYNC_LDS
    if (more) wait_asynccnt<1>();            // in-order: current tile is done
    else      wait_asynccnt<0>();
#endif
    __syncthreads();                          // tile visible to all 8 waves

    // A fragment: 16x32 bf16, lane = row, two contiguous 8-elem runs
    uint4 alo = *(const uint4*)(xrow + kk + 16 * hi);
    uint4 ahi = *(const uint4*)(xrow + kk + 16 * hi + 8);
    v16bf a = mkv16(alo, ahi);
    if (kk + 128 < K) __builtin_prefetch(xrow + kk + 128, 0, 3);

    const unsigned short* wb = &sW[cur * 2048];
#pragma unroll
    for (int t = 0; t < 4; ++t) {
      v16bf b = mkv16u(*(const v8u*)(wb + (t * 16 + ln16) * 32 + 16 * hi));
      c[t] = __builtin_amdgcn_wmma_f32_16x16x32_bf16(false, a, false, b,
                                                     (short)0, c[t], false, false);
    }
    __syncthreads();                          // reads done before overwrite
  }

#pragma unroll
  for (int t = 0; t < 4; ++t) {
#pragma unroll
    for (int r = 0; r < 8; ++r) {
      int rr = m0 + r + 8 * hi;
      int cc = n0 + t * 16 + ln16;
      float v = c[t][r] + bias[cc];
      if (act == 1) v = gelu_exact(v);
      size_t idx = (size_t)rr * N + cc;
      if (gate) v *= silu(gate[idx]);
      if (outF) outF[idx] = v;
      if (outB) outB[idx] = f2bf(v);
    }
  }
}

// ---------------------------------------------------------------- attention
// Causal linear attention (no softmax denominator), per head:
//   o[l,:] = sum_{m<=l} (q[l]·k[m]) * v[m,:]
// One wave per 16-query tile; 32-key chunks; both matmuls via WMMA bf16.
// Scores bounce through per-wave LDS to move from C-layout to A-layout;
// v chunks are staged transposed so B fragments are single 32B ds reads.
__global__ void attn_wmma(const unsigned short* __restrict__ q,
                          const unsigned short* __restrict__ k,
                          const unsigned short* __restrict__ v,
                          float* __restrict__ o) {
  __shared__ __align__(32) unsigned short sS [8 * 16 * 32]; // score tiles, row-major
  __shared__ __align__(32) unsigned short sVT[8 * 32 * 32]; // v chunk, transposed [d][key]

  const int lane = threadIdx.x & 31;
  const int wave = threadIdx.x >> 5;
  const int hi   = lane >> 4;
  const int ln16 = lane & 15;

  const int bh = blockIdx.x;
  const int b  = bh / NH;
  const int h  = bh % NH;
  const int l0 = (blockIdx.y * 8 + wave) * 16;

  const size_t base = (size_t)b * EMB + (size_t)h * HD;
  const unsigned short* qb = q + base;
  const unsigned short* kb = k + base;
  const unsigned short* vb = v + base;

  unsigned short* S  = sS  + wave * (16 * 32);
  unsigned short* VT = sVT + wave * (32 * 32);

  // A fragment of q for this tile (rows l0..l0+15, K = head dim 32)
  const unsigned short* qrow = qb + (size_t)(l0 + ln16) * RS;
  uint4 qlo = *(const uint4*)(qrow + 16 * hi);
  uint4 qhi = *(const uint4*)(qrow + 16 * hi + 8);
  v16bf aq = mkv16(qlo, qhi);

  v8f o0 = {}, o1 = {};
  const v8f zero = {};

  for (int m0 = 0; m0 <= l0 + 15; m0 += 32) {
    // ---- scores (16 q-rows x 32 keys) = aq (16x32) * k^T, two 16x16 WMMA
    v16bf bk0 = mkv16u(*(const v8u*)(kb + (size_t)(m0 +      ln16) * RS + 16 * hi));
    v16bf bk1 = mkv16u(*(const v8u*)(kb + (size_t)(m0 + 16 + ln16) * RS + 16 * hi));
    v8f s0 = __builtin_amdgcn_wmma_f32_16x16x32_bf16(false, aq, false, bk0,
                                                     (short)0, zero, false, false);
    v8f s1 = __builtin_amdgcn_wmma_f32_16x16x32_bf16(false, aq, false, bk1,
                                                     (short)0, zero, false, false);
    // ---- causal mask only on the diagonal chunk
    if (m0 + 31 > l0) {
#pragma unroll
      for (int r = 0; r < 8; ++r) {
        int lq = l0 + r + 8 * hi;
        if (m0 +      ln16 > lq) s0[r] = 0.0f;
        if (m0 + 16 + ln16 > lq) s1[r] = 0.0f;
      }
    }
    // ---- redistribute scores (C layout -> A layout) via LDS, as bf16
#pragma unroll
    for (int r = 0; r < 8; ++r) {
      S[(r + 8 * hi) * 32 +      ln16] = f2bf(s0[r]);
      S[(r + 8 * hi) * 32 + 16 + ln16] = f2bf(s1[r]);
    }
    // ---- stage v chunk transposed: VT[d][key_local]
    {
      const unsigned short* vr = vb + (size_t)(m0 + lane) * RS;
      uint4 p0 = *(const uint4*)(vr +  0);
      uint4 p1 = *(const uint4*)(vr +  8);
      uint4 p2 = *(const uint4*)(vr + 16);
      uint4 p3 = *(const uint4*)(vr + 24);
      unsigned pw[16] = { p0.x,p0.y,p0.z,p0.w, p1.x,p1.y,p1.z,p1.w,
                          p2.x,p2.y,p2.z,p2.w, p3.x,p3.y,p3.z,p3.w };
#pragma unroll
      for (int e = 0; e < 16; ++e) {
        VT[(2 * e    ) * 32 + lane] = (unsigned short)(pw[e] & 0xFFFFu);
        VT[(2 * e + 1) * 32 + lane] = (unsigned short)(pw[e] >> 16);
      }
    }
    // ---- o += scores (16x32) * v (32x32), two 16x16 WMMA
    const unsigned short* srow = S + ln16 * 32;
    uint4 slo = *(const uint4*)(srow + 16 * hi);
    uint4 shi = *(const uint4*)(srow + 16 * hi + 8);
    v16bf as = mkv16(slo, shi);
    v16bf bv0 = mkv16u(*(const v8u*)(VT + (     ln16) * 32 + 16 * hi));
    v16bf bv1 = mkv16u(*(const v8u*)(VT + (16 + ln16) * 32 + 16 * hi));
    o0 = __builtin_amdgcn_wmma_f32_16x16x32_bf16(false, as, false, bv0,
                                                 (short)0, o0, false, false);
    o1 = __builtin_amdgcn_wmma_f32_16x16x32_bf16(false, as, false, bv1,
                                                 (short)0, o1, false, false);
  }

  float* ob = o + base;
#pragma unroll
  for (int r = 0; r < 8; ++r) {
    size_t l = (size_t)(l0 + r + 8 * hi);
    ob[l * RS +      ln16] = o0[r];
    ob[l * RS + 16 + ln16] = o1[r];
  }
}

// ---------------------------------------------------------------- LayerNorm
__global__ void ln_rows(const float* __restrict__ x,
                        const float* __restrict__ g,
                        const float* __restrict__ b,
                        unsigned short* __restrict__ y) {
  __shared__ float rs1[256];
  __shared__ float rs2[256];
  const int row = blockIdx.x;
  const int tid = threadIdx.x;
  const float* xr = x + (size_t)row * EMB;

  float s = 0.f, s2 = 0.f, vals[4];
#pragma unroll
  for (int i = 0; i < 4; ++i) {
    float t = xr[tid + i * 256];
    vals[i] = t; s += t; s2 += t * t;
  }
  rs1[tid] = s; rs2[tid] = s2;
  __syncthreads();
  for (int off = 128; off > 0; off >>= 1) {
    if (tid < off) { rs1[tid] += rs1[tid + off]; rs2[tid] += rs2[tid + off]; }
    __syncthreads();
  }
  const float mu   = rs1[0] * (1.0f / EMB);
  const float var  = rs2[0] * (1.0f / EMB) - mu * mu;
  const float rstd = rsqrtf(var + 1e-5f);
#pragma unroll
  for (int i = 0; i < 4; ++i) {
    int c = tid + i * 256;
    y[(size_t)row * EMB + c] = f2bf((vals[i] - mu) * rstd * g[c] + b[c]);
  }
}

// ---------------------------------------------------------------- launch
extern "C" void kernel_launch(void* const* d_in, const int* in_sizes, int n_in,
                              void* d_out, int out_size, void* d_ws, size_t ws_size,
                              hipStream_t stream) {
  const float* query = (const float*)d_in[0];
  const float* key_  = (const float*)d_in[1];
  const float* value = (const float*)d_in[2];
  const float* Wq  = (const float*)d_in[3];  const float* bq  = (const float*)d_in[4];
  const float* Wk  = (const float*)d_in[5];  const float* bk  = (const float*)d_in[6];
  const float* Wv  = (const float*)d_in[7];  const float* bv  = (const float*)d_in[8];
  const float* Wu  = (const float*)d_in[9];  const float* bu  = (const float*)d_in[10];
  const float* Wo1 = (const float*)d_in[11]; const float* bo1 = (const float*)d_in[12];
  const float* Wo2 = (const float*)d_in[13]; const float* bo2 = (const float*)d_in[14];
  const float* lng = (const float*)d_in[15]; const float* lnb = (const float*)d_in[16];

  const size_t LBE = (size_t)LQ * BBS * EMB;   // 4,194,304
  const size_t EE  = (size_t)EMB * EMB;        // 1,048,576

  char* p = (char*)d_ws;
  auto take = [&](size_t bytes) -> void* {
    void* r = (void*)p;
    p += (bytes + 255) & ~(size_t)255;
    return r;
  };
  unsigned short* qry_b = (unsigned short*)take(LBE * 2);
  unsigned short* key_b = (unsigned short*)take(LBE * 2);
  unsigned short* val_b = (unsigned short*)take(LBE * 2);
  unsigned short* Wq_b  = (unsigned short*)take(EE * 2);
  unsigned short* Wk_b  = (unsigned short*)take(EE * 2);
  unsigned short* Wv_b  = (unsigned short*)take(EE * 2);
  unsigned short* Wu_b  = (unsigned short*)take(EE * 2);
  unsigned short* Wo1_b = (unsigned short*)take(EE * 2);
  unsigned short* Wo2_b = (unsigned short*)take(EE * 2);
  unsigned short* q_b   = (unsigned short*)take(LBE * 2);
  unsigned short* k_b   = (unsigned short*)take(LBE * 2);
  unsigned short* v_b   = (unsigned short*)take(LBE * 2);
  float*          u_f   = (float*)take(LBE * 4);
  float*          o_f   = (float*)take(LBE * 4);
  unsigned short* oln_b = (unsigned short*)take(LBE * 2);
  unsigned short* g_b   = (unsigned short*)take(LBE * 2);

  // --- f32 -> bf16 conversions
  cvt_f32_bf16<<<(int)(LBE / 256), 256, 0, stream>>>(query, qry_b, (int)LBE);
  cvt_f32_bf16<<<(int)(LBE / 256), 256, 0, stream>>>(key_,  key_b, (int)LBE);
  cvt_f32_bf16<<<(int)(LBE / 256), 256, 0, stream>>>(value, val_b, (int)LBE);
  cvt_f32_bf16<<<(int)(EE  / 256), 256, 0, stream>>>(Wq,  Wq_b,  (int)EE);
  cvt_f32_bf16<<<(int)(EE  / 256), 256, 0, stream>>>(Wk,  Wk_b,  (int)EE);
  cvt_f32_bf16<<<(int)(EE  / 256), 256, 0, stream>>>(Wv,  Wv_b,  (int)EE);
  cvt_f32_bf16<<<(int)(EE  / 256), 256, 0, stream>>>(Wu,  Wu_b,  (int)EE);
  cvt_f32_bf16<<<(int)(EE  / 256), 256, 0, stream>>>(Wo1, Wo1_b, (int)EE);
  cvt_f32_bf16<<<(int)(EE  / 256), 256, 0, stream>>>(Wo2, Wo2_b, (int)EE);

  const dim3 gg(MROW / 128, EMB / 64);   // (32, 16)
  // --- projections: q = gelu(x Wq^T + bq), k = gelu(..), v = .., u = ..
  gemm_wmma_bias_act<<<gg, 256, 0, stream>>>(qry_b, Wq_b, bq, nullptr,
                                             nullptr, q_b, MROW, EMB, EMB, 1);
  gemm_wmma_bias_act<<<gg, 256, 0, stream>>>(key_b, Wk_b, bk, nullptr,
                                             nullptr, k_b, MROW, EMB, EMB, 1);
  gemm_wmma_bias_act<<<gg, 256, 0, stream>>>(val_b, Wv_b, bv, nullptr,
                                             nullptr, v_b, MROW, EMB, EMB, 0);
  gemm_wmma_bias_act<<<gg, 256, 0, stream>>>(qry_b, Wu_b, bu, nullptr,
                                             u_f, nullptr, MROW, EMB, EMB, 0);

  // --- causal linear attention (64 heads x 16 q-tile groups, 8 waves/blk)
  attn_wmma<<<dim3(BBS * NH, LQ / 128), 256, 0, stream>>>(q_b, k_b, v_b, o_f);

  // --- LayerNorm over E
  ln_rows<<<MROW, 256, 0, stream>>>(o_f, lng, lnb, oln_b);

  // --- gated output projections
  gemm_wmma_bias_act<<<gg, 256, 0, stream>>>(oln_b, Wo1_b, bo1, u_f,
                                             nullptr, g_b, MROW, EMB, EMB, 0);
  gemm_wmma_bias_act<<<gg, 256, 0, stream>>>(g_b, Wo2_b, bo2, nullptr,
                                             (float*)d_out, nullptr, MROW, EMB, EMB, 0);
}
